// ModifiedGPT2Attention_4930622455889
// MI455X (gfx1250) — matrix operand order repaired and checked by reading, not compiled, not run on hardware
//
#include <hip/hip_runtime.h>

typedef __attribute__((ext_vector_type(16))) __bf16 v16bf;
typedef __attribute__((ext_vector_type(8)))  __bf16 v8bf;
typedef __attribute__((ext_vector_type(8)))  float  v8f;

__device__ __forceinline__ v8f wmma_bf16(v16bf a, v16bf b, v8f c) {
  return __builtin_amdgcn_wmma_f32_16x16x32_bf16(false, a, false, b, (short)0, c, false, false);
}

// A fragment: 16x32 (MxK) bf16 from row-major A (leading dim lda elements).
// ISA layout: lanes 0-15 row m=lane, elems 0..7 -> K=k0..k0+7, elems 8..15 -> K=k0+16..k0+23;
// lanes 16-31 same rows, K offset +8.
__device__ __forceinline__ v16bf load_a_frag(const __bf16* A, int lda, int row0, int k0, int lane) {
  const int m  = lane & 15;
  const int hf = lane >> 4;
  const __bf16* p = A + (size_t)(row0 + m) * lda + k0 + hf * 8;
  v8bf lo = *(const v8bf*)(p);
  v8bf hi = *(const v8bf*)(p + 16);
  return __builtin_shufflevector(lo, hi, 0,1,2,3,4,5,6,7,8,9,10,11,12,13,14,15);
}

// B fragment: 32x16 (KxN) bf16 where memory is Bt[N][K] row-major (ldk = K stride).
// ISA layout: lane holds column n=lane&15; elems 0..7 -> K=k0+8*hf.., elems 8..15 -> K=k0+16+8*hf.. (same as A).
__device__ __forceinline__ v16bf load_bt_frag(const __bf16* Bt, int ldk, int col0, int k0, int lane) {
  const int n  = lane & 15;
  const int hf = lane >> 4;
  const __bf16* p = Bt + (size_t)(col0 + n) * ldk + k0 + hf * 8;
  v8bf lo = *(const v8bf*)(p);
  v8bf hi = *(const v8bf*)(p + 16);
  return __builtin_shufflevector(lo, hi, 0,1,2,3,4,5,6,7,8,9,10,11,12,13,14,15);
}

// ---------------- conversion kernels ----------------
__global__ void cvt_bf16_kernel(const float* __restrict__ src, __bf16* __restrict__ dst, int n) {
  int i = blockIdx.x * blockDim.x + threadIdx.x;
  if (i < n) dst[i] = (__bf16)src[i];
}

// src [rows][cols] f32 -> dst [cols][rows] bf16
__global__ void cvt_transpose_kernel(const float* __restrict__ src, __bf16* __restrict__ dst,
                                     int rows, int cols) {
  int i = blockIdx.x * blockDim.x + threadIdx.x;
  if (i < rows * cols) {
    int r = i / cols, c = i % cols;
    dst[(size_t)c * rows + r] = (__bf16)src[i];
  }
}

// ---------------- QKV projection ----------------
// X[4096][1024] bf16 @ Wt[3072][1024] (transposed) + bias -> scatter into
// Q[B,H,S,hd] (pre-scaled by 0.125), K[B,H,S,hd], Vt[B,H,hd,S].
// One wave computes a 32x64 output tile: 2 A-frags x 4 B-frags -> 8 WMMAs per k-step.
__global__ void __launch_bounds__(256, 1)
qkv_gemm_kernel(const __bf16* __restrict__ X, const __bf16* __restrict__ Wt,
                const float* __restrict__ bias,
                __bf16* __restrict__ Qb, __bf16* __restrict__ Kb,
                __bf16* __restrict__ Vtb) {
  const int wave = threadIdx.x >> 5, lane = threadIdx.x & 31;
  const int tile = blockIdx.x * 8 + wave;           // 6144 waves total
  const int tm = tile / 48, tn4 = tile % 48;        // 128 row tiles x 48 col-strips
  const int row0 = tm * 32, col0 = tn4 * 64;
  const int nl = lane & 15, hf = lane >> 4;

  v8f acc[2][4];
#pragma unroll
  for (int g = 0; g < 2; g++)
#pragma unroll
    for (int t = 0; t < 4; t++) acc[g][t] = v8f{};

#pragma unroll 2
  for (int k0 = 0; k0 < 1024; k0 += 32) {
    v16bf a0 = load_a_frag(X, 1024, row0,      k0, lane);
    v16bf a1 = load_a_frag(X, 1024, row0 + 16, k0, lane);
    v16bf b0 = load_bt_frag(Wt, 1024, col0,      k0, lane);
    v16bf b1 = load_bt_frag(Wt, 1024, col0 + 16, k0, lane);
    v16bf b2 = load_bt_frag(Wt, 1024, col0 + 32, k0, lane);
    v16bf b3 = load_bt_frag(Wt, 1024, col0 + 48, k0, lane);
    acc[0][0] = wmma_bf16(a0, b0, acc[0][0]);
    acc[0][1] = wmma_bf16(a0, b1, acc[0][1]);
    acc[0][2] = wmma_bf16(a0, b2, acc[0][2]);
    acc[0][3] = wmma_bf16(a0, b3, acc[0][3]);
    acc[1][0] = wmma_bf16(a1, b0, acc[1][0]);
    acc[1][1] = wmma_bf16(a1, b1, acc[1][1]);
    acc[1][2] = wmma_bf16(a1, b2, acc[1][2]);
    acc[1][3] = wmma_bf16(a1, b3, acc[1][3]);
  }

#pragma unroll
  for (int t = 0; t < 4; t++) {
    int col = col0 + t * 16 + nl;                   // 0..3071
    float bv = bias[col];
    int which = col >> 10, d = col & 1023;
    int h = d >> 6, di = d & 63;
#pragma unroll
    for (int g = 0; g < 2; g++) {
#pragma unroll
      for (int r = 0; r < 8; r++) {
        int row = row0 + g * 16 + r + 8 * hf;       // 0..4095
        int bb = row >> 11, s = row & 2047;
        int bh = bb * 16 + h;
        float val = acc[g][t][r] + bv;
        if (which == 0)      Qb[((size_t)bh * 2048 + s) * 64 + di] = (__bf16)(val * 0.125f);
        else if (which == 1) Kb[((size_t)bh * 2048 + s) * 64 + di] = (__bf16)val;
        else                 Vtb[((size_t)bh * 64 + di) * 2048 + s] = (__bf16)val;
      }
    }
  }
}

// ---------------- flash attention ----------------
// One wave per 16-query tile per (b,h). Causal, online softmax, f32 accumulators.
__global__ void __launch_bounds__(256, 1)
flash_attn_kernel(const __bf16* __restrict__ Q, const __bf16* __restrict__ K,
                  const __bf16* __restrict__ Vt, __bf16* __restrict__ Aout) {
  __shared__ __align__(16) __bf16 Plds[8][16][32];  // per-wave 16x32 P staging
  const int wave = threadIdx.x >> 5, lane = threadIdx.x & 31;
  const int nl = lane & 15, hf = lane >> 4;
  const int gw = blockIdx.x * 8 + wave;             // 4096 waves
  const int bh = gw >> 7, qt = gw & 127;
  const int q0 = qt * 16;
  const __bf16* Qh = Q + (size_t)bh * 2048 * 64;
  const __bf16* Kh = K + (size_t)bh * 2048 * 64;
  const __bf16* Vh = Vt + (size_t)bh * 64 * 2048;

  // Q fragments for the full hd=64 contraction, kept in registers.
  v16bf qa0 = load_a_frag(Qh, 64, q0, 0, lane);
  v16bf qa1 = load_a_frag(Qh, 64, q0, 32, lane);

  float mrow[8], lrow[8];
  v8f o0 = {}, o1 = {}, o2 = {}, o3 = {};
#pragma unroll
  for (int r = 0; r < 8; r++) { mrow[r] = -3.0e38f; lrow[r] = 0.0f; }

  const int nkb = ((q0 + 15) >> 5) + 1;             // key blocks of 32 (causal bound)
  for (int kb = 0; kb < nkb; kb++) {
    const int k0 = kb * 32;

    // Hint the next key block into cache while this block's serial chain runs.
    if (kb + 1 < nkb) {
      const __bf16* nK = Kh + (size_t)(k0 + 32) * 64;   // next 4KB K block
      __builtin_prefetch(nK, 0, 1);
      __builtin_prefetch(nK + 1024, 0, 1);
      __builtin_prefetch(nK + 2048, 0, 1);
      __builtin_prefetch(nK + 3072, 0, 1);
    }

    // scores: Q[16x64] @ K^T -> two 16x16 tiles (keys k0..+15, k0+16..+31)
    v8f s0 = {}, s1 = {};
    s0 = wmma_bf16(qa0, load_bt_frag(Kh, 64, k0,      0, lane), s0);
    s1 = wmma_bf16(qa0, load_bt_frag(Kh, 64, k0 + 16, 0, lane), s1);
    s0 = wmma_bf16(qa1, load_bt_frag(Kh, 64, k0,     32, lane), s0);
    s1 = wmma_bf16(qa1, load_bt_frag(Kh, 64, k0 + 16, 32, lane), s1);

#pragma unroll
    for (int r = 0; r < 8; r++) {
      const int m = r + 8 * hf;
      const int qidx = q0 + m;
      float sa = (k0 + nl      <= qidx) ? s0[r] : -3.0e38f;   // causal mask (Q pre-scaled)
      float sb = (k0 + 16 + nl <= qidx) ? s1[r] : -3.0e38f;
      float mx = fmaxf(sa, sb);
      mx = fmaxf(mx, __shfl_xor(mx, 1));
      mx = fmaxf(mx, __shfl_xor(mx, 2));
      mx = fmaxf(mx, __shfl_xor(mx, 4));
      mx = fmaxf(mx, __shfl_xor(mx, 8));            // row max within 16-lane half-group
      float mnew = fmaxf(mrow[r], mx);
      float corr = __expf(mrow[r] - mnew);
      float p0 = __expf(sa - mnew);
      float p1 = __expf(sb - mnew);
      float ps = p0 + p1;
      ps += __shfl_xor(ps, 1);
      ps += __shfl_xor(ps, 2);
      ps += __shfl_xor(ps, 4);
      ps += __shfl_xor(ps, 8);                      // row sum
      lrow[r] = lrow[r] * corr + ps;
      mrow[r] = mnew;
      o0[r] *= corr; o1[r] *= corr; o2[r] *= corr; o3[r] *= corr;
      Plds[wave][m][nl]      = (__bf16)p0;          // C-layout -> LDS
      Plds[wave][m][nl + 16] = (__bf16)p1;
    }
    __builtin_amdgcn_wave_barrier();
    asm volatile("" ::: "memory");                  // LDS in-order within wave

    // Re-read P as an A-fragment (16 queries x 32 keys).
    v8bf plo = *(const v8bf*)(&Plds[wave][nl][hf * 8]);
    v8bf phi = *(const v8bf*)(&Plds[wave][nl][16 + hf * 8]);
    v16bf pa = __builtin_shufflevector(plo, phi, 0,1,2,3,4,5,6,7,8,9,10,11,12,13,14,15);

    o0 = wmma_bf16(pa, load_bt_frag(Vh, 2048,  0, k0, lane), o0);
    o1 = wmma_bf16(pa, load_bt_frag(Vh, 2048, 16, k0, lane), o1);
    o2 = wmma_bf16(pa, load_bt_frag(Vh, 2048, 32, k0, lane), o2);
    o3 = wmma_bf16(pa, load_bt_frag(Vh, 2048, 48, k0, lane), o3);
    __builtin_amdgcn_wave_barrier();
    asm volatile("" ::: "memory");
  }

  const int b = bh >> 4, h = bh & 15;
#pragma unroll
  for (int r = 0; r < 8; r++) {
    int s = q0 + r + 8 * hf;
    float inv = 1.0f / lrow[r];
    size_t base = ((size_t)b * 2048 + s) * 1024 + h * 64 + nl;   // merged-heads [B,S,D]
    Aout[base + 0]  = (__bf16)(o0[r] * inv);
    Aout[base + 16] = (__bf16)(o1[r] * inv);
    Aout[base + 32] = (__bf16)(o2[r] * inv);
    Aout[base + 48] = (__bf16)(o3[r] * inv);
  }
}

// ---------------- output projection ----------------
// One wave computes a 32x64 tile; f32 output + bias.
__global__ void __launch_bounds__(256, 1)
proj_gemm_kernel(const __bf16* __restrict__ A, const __bf16* __restrict__ Wt,
                 const float* __restrict__ bias, float* __restrict__ out) {
  const int wave = threadIdx.x >> 5, lane = threadIdx.x & 31;
  const int tile = blockIdx.x * 8 + wave;           // 2048 waves
  const int tm = tile / 16, tn4 = tile % 16;        // 128 x 16 (64-wide strips)
  const int row0 = tm * 32, col0 = tn4 * 64;
  const int nl = lane & 15, hf = lane >> 4;

  v8f acc[2][4];
#pragma unroll
  for (int g = 0; g < 2; g++)
#pragma unroll
    for (int t = 0; t < 4; t++) acc[g][t] = v8f{};

#pragma unroll 2
  for (int k0 = 0; k0 < 1024; k0 += 32) {
    v16bf a0 = load_a_frag(A, 1024, row0,      k0, lane);
    v16bf a1 = load_a_frag(A, 1024, row0 + 16, k0, lane);
    v16bf b0 = load_bt_frag(Wt, 1024, col0,      k0, lane);
    v16bf b1 = load_bt_frag(Wt, 1024, col0 + 16, k0, lane);
    v16bf b2 = load_bt_frag(Wt, 1024, col0 + 32, k0, lane);
    v16bf b3 = load_bt_frag(Wt, 1024, col0 + 48, k0, lane);
    acc[0][0] = wmma_bf16(a0, b0, acc[0][0]);
    acc[0][1] = wmma_bf16(a0, b1, acc[0][1]);
    acc[0][2] = wmma_bf16(a0, b2, acc[0][2]);
    acc[0][3] = wmma_bf16(a0, b3, acc[0][3]);
    acc[1][0] = wmma_bf16(a1, b0, acc[1][0]);
    acc[1][1] = wmma_bf16(a1, b1, acc[1][1]);
    acc[1][2] = wmma_bf16(a1, b2, acc[1][2]);
    acc[1][3] = wmma_bf16(a1, b3, acc[1][3]);
  }

#pragma unroll
  for (int t = 0; t < 4; t++) {
    int col = col0 + t * 16 + nl;
    float bv = bias[col];
#pragma unroll
    for (int g = 0; g < 2; g++)
#pragma unroll
      for (int r = 0; r < 8; r++)
        out[(size_t)(row0 + g * 16 + r + 8 * hf) * 1024 + col] = acc[g][t][r] + bv;
  }
}

extern "C" void kernel_launch(void* const* d_in, const int* in_sizes, int n_in,
                              void* d_out, int out_size, void* d_ws, size_t ws_size,
                              hipStream_t stream) {
  const float* hidden = (const float*)d_in[0];   // [2,2048,1024]
  const float* w_attn = (const float*)d_in[1];   // [1024,3072]
  const float* b_attn = (const float*)d_in[2];   // [3072]
  const float* w_proj = (const float*)d_in[3];   // [1024,1024]
  const float* b_proj = (const float*)d_in[4];   // [1024]
  float* out = (float*)d_out;

  char* ws = (char*)d_ws;
  __bf16* X      = (__bf16*)(ws);                          //  8 MB: 4096x1024
  __bf16* Wqkvt  = (__bf16*)(ws + (8u << 20));             //  6 MB: 3072x1024 (transposed)
  __bf16* Wprojt = (__bf16*)(ws + (14u << 20));            //  2 MB: 1024x1024 (transposed)
  __bf16* Qb     = (__bf16*)(ws + (16u << 20));            //  8 MB: [32][2048][64]
  __bf16* Kb     = (__bf16*)(ws + (24u << 20));            //  8 MB: [32][2048][64]
  __bf16* Vtb    = (__bf16*)(ws + (32u << 20));            //  8 MB: [32][64][2048]
  __bf16* Aout   = (__bf16*)(ws + (40u << 20));            //  8 MB: [2,2048,1024]

  cvt_bf16_kernel<<<(4096 * 1024) / 256, 256, 0, stream>>>(hidden, X, 4096 * 1024);
  cvt_transpose_kernel<<<(1024 * 3072) / 256, 256, 0, stream>>>(w_attn, Wqkvt, 1024, 3072);
  cvt_transpose_kernel<<<(1024 * 1024) / 256, 256, 0, stream>>>(w_proj, Wprojt, 1024, 1024);

  qkv_gemm_kernel<<<768, 256, 0, stream>>>(X, Wqkvt, b_attn, Qb, Kb, Vtb);
  flash_attn_kernel<<<512, 256, 0, stream>>>(Qb, Kb, Vtb, Aout);
  proj_gemm_kernel<<<256, 256, 0, stream>>>(Aout, Wprojt, b_proj, out);
}
